// GIN_26912265077021
// MI455X (gfx1250) — matrix operand order repaired and checked by reading, not compiled, not run on hardware
//
#include <hip/hip_runtime.h>

// GIN (3-layer, eval-mode BN) for MI455X / gfx1250, wave32 + WMMA.
// Aggregation via per-call CSR build + gather (no float atomics on the hot
// path); fused GEMM1->BN->ReLU->GEMM2->ReLU per layer with f16 WMMA
// (f32 accumulate); tiny final MLP also on WMMA.
#define N_NODES 50000
#define N_EDGES 800000
#define HID     128
#define OUTD    64
#define NLAYER  3
#define NGRAPH  256
#define BN_EPS  1e-5f

typedef __attribute__((ext_vector_type(16))) _Float16 v16h;
typedef __attribute__((ext_vector_type(8)))  float    v8f;

// LDS row stride in halves (128 + 8 pad) -> 272B rows, keeps 16B alignment
// for b128 LDS accesses and staggers banks across rows.
#define LDSW 136

// ---------------------------------------------------------------------------
// Utility kernels
// ---------------------------------------------------------------------------
__global__ __launch_bounds__(256) void zero_f4(float4* __restrict__ d, int n4) {
    int i = blockIdx.x * 256 + threadIdx.x;
    if (i < n4) d[i] = make_float4(0.f, 0.f, 0.f, 0.f);
}

__global__ __launch_bounds__(256) void zero_i(int* __restrict__ d, int n) {
    int i = blockIdx.x * 256 + threadIdx.x;
    if (i < n) d[i] = 0;
}

__global__ __launch_bounds__(256) void copy_i(const int* __restrict__ s,
                                              int* __restrict__ d, int n) {
    int i = blockIdx.x * 256 + threadIdx.x;
    if (i < n) d[i] = s[i];
}

// ---------------------------------------------------------------------------
// CSR build (once per call; edge_index is reused by all 3 layers).
// ---------------------------------------------------------------------------
__global__ __launch_bounds__(256) void deg_count(const int* __restrict__ eidx,
                                                 int* __restrict__ deg) {
    int e = blockIdx.x * 256 + threadIdx.x;
    if (e < N_EDGES) atomicAdd(&deg[eidx[N_EDGES + e]], 1);
}

// Single-block exclusive scan: rowptr[0..n] from deg[0..n-1].
__global__ __launch_bounds__(256) void scan_excl(const int* __restrict__ deg,
                                                 int* __restrict__ rowptr,
                                                 int n) {
    __shared__ int sh[256];
    __shared__ int carry;
    if (threadIdx.x == 0) carry = 0;
    __syncthreads();
    for (int base = 0; base < n; base += 256) {
        int i = base + threadIdx.x;
        int v = (i < n) ? deg[i] : 0;
        sh[threadIdx.x] = v;
        __syncthreads();
#pragma unroll
        for (int off = 1; off < 256; off <<= 1) {
            int t = (threadIdx.x >= off) ? sh[threadIdx.x - off] : 0;
            __syncthreads();
            sh[threadIdx.x] += t;
            __syncthreads();
        }
        if (i < n) rowptr[i] = carry + sh[threadIdx.x] - v;  // exclusive
        __syncthreads();
        if (threadIdx.x == 0) carry += sh[255];
        __syncthreads();
    }
    if (threadIdx.x == 0) rowptr[n] = carry;  // == N_EDGES
}

__global__ __launch_bounds__(256) void fill_csr(const int* __restrict__ eidx,
                                                int* __restrict__ cursor,
                                                int* __restrict__ col) {
    int e = blockIdx.x * 256 + threadIdx.x;
    if (e >= N_EDGES) return;
    int d = eidx[N_EDGES + e];
    int pos = atomicAdd(&cursor[d], 1);
    col[pos] = eidx[e];
}

// ---------------------------------------------------------------------------
// Gather: agg[n] = x[n] + sum_{j in adj(n)} x[j]. One wave per node; each
// lane owns 4 consecutive features (float4). All row reads are 512B,
// coalesced, L2-resident. Neighbor loop unrolled x2 to keep two independent
// row loads in flight behind the col[k] index chain.
// ---------------------------------------------------------------------------
__global__ __launch_bounds__(256) void gin_gather(const float* __restrict__ x,
                                                  const int* __restrict__ rowptr,
                                                  const int* __restrict__ col,
                                                  float* __restrict__ agg) {
    int t = blockIdx.x * 256 + threadIdx.x;
    int node = t >> 5;
    if (node >= N_NODES) return;
    int c = (threadIdx.x & 31) << 2;

    float4 acc = *(const float4*)(x + (size_t)node * HID + c);
    int k   = rowptr[node];
    int end = rowptr[node + 1];
    for (; k + 1 < end; k += 2) {
        int j0 = col[k], j1 = col[k + 1];
        float4 v0 = *(const float4*)(x + (size_t)j0 * HID + c);
        float4 v1 = *(const float4*)(x + (size_t)j1 * HID + c);
        acc.x += v0.x + v1.x; acc.y += v0.y + v1.y;
        acc.z += v0.z + v1.z; acc.w += v0.w + v1.w;
    }
    if (k < end) {
        int j = col[k];
        float4 v = *(const float4*)(x + (size_t)j * HID + c);
        acc.x += v.x; acc.y += v.y; acc.z += v.z; acc.w += v.w;
    }
    *(float4*)(agg + (size_t)node * HID + c) = acc;
}

// pooled[batch[n]] += x[n]  (small: 6.4M f32 atomics, one pass)
__global__ __launch_bounds__(256) void gin_pool(const float* __restrict__ x,
                                                const int* __restrict__ batch,
                                                float* __restrict__ pooled) {
    int t = blockIdx.x * 256 + threadIdx.x;
    int n = t >> 5;
    if (n >= N_NODES) return;
    int c = (t & 31) << 2;
    int g = batch[n];
    float4 v = *(const float4*)(x + (size_t)n * HID + c);
    float* p = pooled + (size_t)g * HID + c;
    unsafeAtomicAdd(p + 0, v.x);
    unsafeAtomicAdd(p + 1, v.y);
    unsafeAtomicAdd(p + 2, v.z);
    unsafeAtomicAdd(p + 3, v.w);
}

// ---------------------------------------------------------------------------
// Fragment helpers (wave32 WMMA f16 layouts per CDNA5 ISA 7.12.2):
//  A (16x32, f16): lane L: M = L%16, hi = L/16;
//    halves 0..7  = K = 8*hi + 0..7, halves 8..15 = K = 16 + 8*hi + 0..7.
//  B (32x16, f16): lane L: N = L%16, hi = L/16; same K order as A.
//  C/D (16x16 f32): lane L: N = L%16; vgpr v: M = v + 8*(L/16).
// ---------------------------------------------------------------------------
__device__ inline v16h load_a_frag(const _Float16* __restrict__ smem,
                                   int m, int hi, int kc) {
    const _Float16* base = smem + m * LDSW + kc * 32 + hi * 8;
    v16h a;
    ((uint4*)&a)[0] = *(const uint4*)(base);
    ((uint4*)&a)[1] = *(const uint4*)(base + 16);
    return a;
}

__device__ inline v16h load_b_frag(const float* __restrict__ W, int ld,
                                   int col, int hi, int kc) {
    v16h b;
    int kbase = kc * 32 + hi * 8;
#pragma unroll
    for (int j = 0; j < 16; ++j)
        b[j] = (_Float16)W[(size_t)(kbase + (j < 8 ? j : j + 8)) * ld + col];
    return b;
}

// Prefetch all 4 A fragments (8x ds_load_b128), then run the WMMA chain so
// the compiler can use progressive s_wait_dscnt instead of wait-0 per WMMA.
__device__ inline v8f gemm_k128(const _Float16* __restrict__ smem, int m,
                                int hi, const v16h wf[4]) {
    v16h a0 = load_a_frag(smem, m, hi, 0);
    v16h a1 = load_a_frag(smem, m, hi, 1);
    v16h a2 = load_a_frag(smem, m, hi, 2);
    v16h a3 = load_a_frag(smem, m, hi, 3);
    v8f c = {};
    c = __builtin_amdgcn_wmma_f32_16x16x32_f16(false, a0, false, wf[0],
                                               (short)0, c, false, false);
    c = __builtin_amdgcn_wmma_f32_16x16x32_f16(false, a1, false, wf[1],
                                               (short)0, c, false, false);
    c = __builtin_amdgcn_wmma_f32_16x16x32_f16(false, a2, false, wf[2],
                                               (short)0, c, false, false);
    c = __builtin_amdgcn_wmma_f32_16x16x32_f16(false, a3, false, wf[3],
                                               (short)0, c, false, false);
    return c;
}

// ---------------------------------------------------------------------------
// One GIN layer: out = relu( relu(BN(agg @ W1 + b1)) @ W2 + b2 )
// agg already holds x + sum_neighbors. 8 waves/block; wave w owns output
// columns [16w, 16w+16). Grid-stride over 16-row node tiles so weight
// fragments (in registers) are reused across tiles.
// ---------------------------------------------------------------------------
__global__ __launch_bounds__(256) void gin_layer(
    const float* __restrict__ agg, const float* __restrict__ W1,
    const float* __restrict__ W2, const float* __restrict__ b1,
    const float* __restrict__ b2, const float* __restrict__ beta,
    const float* __restrict__ gamma, const float* __restrict__ rm,
    const float* __restrict__ rv, float* __restrict__ out, int n_tiles) {
    __shared__ alignas(16) _Float16 sA[16 * LDSW];
    __shared__ alignas(16) _Float16 sH[16 * LDSW];

    const int lane = threadIdx.x & 31;
    const int wave = threadIdx.x >> 5;  // output col tile 0..7
    const int m    = lane & 15;
    const int hi   = lane >> 4;
    const int col  = wave * 16 + m;

    // Weight fragments, loaded once per block (L2-resident).
    v16h w1f[4], w2f[4];
#pragma unroll
    for (int kc = 0; kc < 4; ++kc) {
        w1f[kc] = load_b_frag(W1, HID, col, hi, kc);
        w2f[kc] = load_b_frag(W2, HID, col, hi, kc);
    }
    // Fold bias + eval-mode BN into per-column scale/offset:
    // BN(h+b1) = h*sc + of, sc = gamma*rsqrt(rv+eps), of = (b1-rm)*sc + beta
    const float sc    = gamma[col] * rsqrtf(rv[col] + BN_EPS);
    const float of    = (b1[col] - rm[col]) * sc + beta[col];
    const float bias2 = b2[col];

    for (int tile = blockIdx.x; tile < n_tiles; tile += gridDim.x) {
        // Stage A tile (16 x 128 f32 -> f16 LDS), 8 floats per thread.
        {
            const int r = threadIdx.x >> 4;
            const int c = (threadIdx.x & 15) * 8;
            const float4* p =
                (const float4*)(agg + ((size_t)tile * 16 + r) * HID + c);
            float4 a0 = p[0], a1 = p[1];
            union { uint4 u; _Float16 h[8]; } tmp;
            tmp.h[0] = (_Float16)a0.x; tmp.h[1] = (_Float16)a0.y;
            tmp.h[2] = (_Float16)a0.z; tmp.h[3] = (_Float16)a0.w;
            tmp.h[4] = (_Float16)a1.x; tmp.h[5] = (_Float16)a1.y;
            tmp.h[6] = (_Float16)a1.z; tmp.h[7] = (_Float16)a1.w;
            *(uint4*)&sA[r * LDSW + c] = tmp.u;
        }
        __syncthreads();

        // GEMM1 + BN + ReLU epilogue -> sH (f16)
        v8f c1 = gemm_k128(sA, m, hi, w1f);
#pragma unroll
        for (int v = 0; v < 8; ++v) {
            float h = c1[v] * sc + of;
            h = h > 0.f ? h : 0.f;
            sH[(v + 8 * hi) * LDSW + col] = (_Float16)h;
        }
        __syncthreads();

        // GEMM2 + bias + ReLU -> out
        v8f c2 = gemm_k128(sH, m, hi, w2f);
#pragma unroll
        for (int v = 0; v < 8; ++v) {
            float h = c2[v] + bias2;
            h = h > 0.f ? h : 0.f;
            out[((size_t)tile * 16 + v + 8 * hi) * HID + col] = h;
        }
        __syncthreads();
    }
}

// ---------------------------------------------------------------------------
// Final MLP on pooled graph features: out = relu(g @ W1 + b1) @ W2 + b2
// 16 blocks (G/16 row tiles); waves 0..7 do GEMM1, waves 0..3 do GEMM2.
// ---------------------------------------------------------------------------
__global__ __launch_bounds__(256) void gin_final(
    const float* __restrict__ g, const float* __restrict__ W1,
    const float* __restrict__ b1, const float* __restrict__ W2,
    const float* __restrict__ b2, float* __restrict__ out) {
    __shared__ alignas(16) _Float16 sA[16 * LDSW];
    __shared__ alignas(16) _Float16 sH[16 * LDSW];

    const int lane = threadIdx.x & 31;
    const int wave = threadIdx.x >> 5;
    const int m    = lane & 15;
    const int hi   = lane >> 4;
    const int tile = blockIdx.x;  // 0..15

    {
        const int r = threadIdx.x >> 4;
        const int c = (threadIdx.x & 15) * 8;
        const float4* p = (const float4*)(g + ((size_t)tile * 16 + r) * HID + c);
        float4 a0 = p[0], a1 = p[1];
        union { uint4 u; _Float16 h[8]; } tmp;
        tmp.h[0] = (_Float16)a0.x; tmp.h[1] = (_Float16)a0.y;
        tmp.h[2] = (_Float16)a0.z; tmp.h[3] = (_Float16)a0.w;
        tmp.h[4] = (_Float16)a1.x; tmp.h[5] = (_Float16)a1.y;
        tmp.h[6] = (_Float16)a1.z; tmp.h[7] = (_Float16)a1.w;
        *(uint4*)&sA[r * LDSW + c] = tmp.u;
    }
    __syncthreads();

    {   // GEMM1 over all 8 col tiles
        const int col = wave * 16 + m;
        v16h bf[4];
#pragma unroll
        for (int kc = 0; kc < 4; ++kc) bf[kc] = load_b_frag(W1, HID, col, hi, kc);
        v8f c1 = gemm_k128(sA, m, hi, bf);
        const float bb = b1[col];
#pragma unroll
        for (int v = 0; v < 8; ++v) {
            float h = c1[v] + bb;
            h = h > 0.f ? h : 0.f;
            sH[(v + 8 * hi) * LDSW + col] = (_Float16)h;
        }
    }
    __syncthreads();

    if (wave < 4) {  // OUTD = 64 -> 4 col tiles; branch is wave-uniform
        const int col = wave * 16 + m;
        v16h bf[4];
#pragma unroll
        for (int kc = 0; kc < 4; ++kc) bf[kc] = load_b_frag(W2, OUTD, col, hi, kc);
        v8f c2 = gemm_k128(sH, m, hi, bf);
        const float bb = b2[col];
#pragma unroll
        for (int v = 0; v < 8; ++v)
            out[((size_t)tile * 16 + v + 8 * hi) * OUTD + col] = c2[v] + bb;
    }
}

// ---------------------------------------------------------------------------
// Launch. Inputs (flat, JAX pytree / alphabetical key order):
//   [0] x [N*128] f32, [1] edge_index [2*E] i32, [2] batch [N] i32,
//   [3 + 8l + {0..7}] conv l: W1,W2,b1,b2,beta,gamma,rm,rv
//   [27..30] final: W1,W2,b1,b2
// Workspace: agg [N*128] f32 | xbuf [N*128] f32 | pooled [G*128] f32 |
//            deg [N] | rowptr [N+1] | cursor [N] | col [E]   (~55 MB)
// ---------------------------------------------------------------------------
extern "C" void kernel_launch(void* const* d_in, const int* in_sizes, int n_in,
                              void* d_out, int out_size, void* d_ws,
                              size_t ws_size, hipStream_t stream) {
    const float* x     = (const float*)d_in[0];
    const int*   eidx  = (const int*)d_in[1];
    const int*   batch = (const int*)d_in[2];

    const float* cw[NLAYER][8];
    for (int l = 0; l < NLAYER; ++l)
        for (int j = 0; j < 8; ++j)
            cw[l][j] = (const float*)d_in[3 + l * 8 + j];
    const float* fW1 = (const float*)d_in[3 + 8 * NLAYER + 0];
    const float* fW2 = (const float*)d_in[3 + 8 * NLAYER + 1];
    const float* fb1 = (const float*)d_in[3 + 8 * NLAYER + 2];
    const float* fb2 = (const float*)d_in[3 + 8 * NLAYER + 3];

    float* agg    = (float*)d_ws;
    float* xbuf   = agg + (size_t)N_NODES * HID;
    float* pooled = xbuf + (size_t)N_NODES * HID;
    int*   deg    = (int*)(pooled + (size_t)NGRAPH * HID);
    int*   rowptr = deg + N_NODES;
    int*   cursor = rowptr + N_NODES + 1;
    int*   col    = cursor + N_NODES;

    const int n_tiles = N_NODES / 16;  // 3125 exact
    const int eb = (N_EDGES + 255) / 256;
    const int nb = (N_NODES + 255) / 256;

    // ---- CSR build (edge_index reused by all layers) ----
    zero_i<<<nb, 256, 0, stream>>>(deg, N_NODES);
    deg_count<<<eb, 256, 0, stream>>>(eidx, deg);
    scan_excl<<<1, 256, 0, stream>>>(deg, rowptr, N_NODES);
    copy_i<<<nb, 256, 0, stream>>>(rowptr, cursor, N_NODES);
    fill_csr<<<eb, 256, 0, stream>>>(eidx, cursor, col);

    // ---- layers ----
    const float* xin = x;
    for (int l = 0; l < NLAYER; ++l) {
        gin_gather<<<(N_NODES * 32) / 256, 256, 0, stream>>>(xin, rowptr, col,
                                                             agg);
        gin_layer<<<512, 256, 0, stream>>>(
            agg, cw[l][0], cw[l][1], cw[l][2], cw[l][3], cw[l][4], cw[l][5],
            cw[l][6], cw[l][7], xbuf, n_tiles);
        xin = xbuf;
    }

    // ---- pool + final MLP ----
    const int pool4 = NGRAPH * HID / 4;
    zero_f4<<<(pool4 + 255) / 256, 256, 0, stream>>>((float4*)pooled, pool4);
    gin_pool<<<(N_NODES * 32) / 256, 256, 0, stream>>>(xin, batch, pooled);
    gin_final<<<NGRAPH / 16, 256, 0, stream>>>(pooled, fW1, fb1, fW2, fb2,
                                               (float*)d_out);
}
